// LearnR_79190607004101
// MI455X (gfx1250) — matrix operand order repaired and checked by reading, not compiled, not run on hardware
//
#include <hip/hip_runtime.h>
#include <cstdint>

// ---------- types ----------
typedef __bf16 bf16_t;
typedef bf16_t v16bf __attribute__((ext_vector_type(16)));
typedef float  v8f   __attribute__((ext_vector_type(8)));

union FragBF { unsigned int u[8]; v16bf v; };

#define NR   8192
#define H2D  128

// ---------- CDNA5 async-copy availability (compile-time safe) ----------
#if defined(__has_builtin)
#  if __has_builtin(__builtin_amdgcn_global_load_async_to_lds_b128)
#    define HAVE_ASYNC_LDS 1
#  endif
#  if __has_builtin(__builtin_amdgcn_s_wait_asynccnt)
#    define HAVE_WAIT_ASYNC 1
#  endif
#endif
#ifndef HAVE_ASYNC_LDS
#  define HAVE_ASYNC_LDS 0
#endif
#ifndef HAVE_WAIT_ASYNC
#  define HAVE_WAIT_ASYNC 0
#endif

// builtin signature (from hipcc diagnostic): first arg is
//   int __attribute__((vector_size(16))) __device__ *   (addrspace(1))
typedef int i4v __attribute__((vector_size(16)));
typedef __attribute__((address_space(1))) i4v g_i4;
typedef __attribute__((address_space(3))) i4v l_i4;

#if HAVE_ASYNC_LDS
__device__ __forceinline__ void async_b128(const void* g, void* l) {
    __builtin_amdgcn_global_load_async_to_lds_b128((g_i4*)g, (l_i4*)l, 0, 0);
}
#endif

__device__ __forceinline__ void wait_async0() {
#if HAVE_WAIT_ASYNC
    __builtin_amdgcn_s_wait_asynccnt(0);
#else
    asm volatile("s_wait_asynccnt 0" ::: "memory");
#endif
}

__device__ __forceinline__ unsigned short f2bf(float f) {
    unsigned int u = __float_as_uint(f);
    u += 0x7FFFu + ((u >> 16) & 1u);           // round-to-nearest-even
    return (unsigned short)(u >> 16);
}

// ---------------------------------------------------------------------------
// Generic NN GEMM: C[M,N] = (relu?)(A[M,K] @ B[K,N]), fp32 in/out, bf16 WMMA.
// Block tile 64(M) x 128(N), BK=32, 256 threads = 8 waves.
// Optionally also emits a bf16 copy of C (used to feed the recon passes).
// ---------------------------------------------------------------------------
__global__ void __launch_bounds__(256)
gemm_nn_bf16(const float* __restrict__ A, int lda,
             const float* __restrict__ B, int ldb,
             float* __restrict__ C, int ldc,
             int K, int relu, unsigned short* __restrict__ Cbf)
{
    __shared__ unsigned short As[64 * 34];    // [m][k], stride 34 (dword-aligned rows)
    __shared__ unsigned short Bs[128 * 34];   // transposed: [n][k]

    const int tid  = threadIdx.x;
    const int lane = tid & 31;
    const int w    = tid >> 5;
    const int m0   = blockIdx.x * 64;
    const int n0   = blockIdx.y * 128;

    const int ms = (w & 3) * 16;
    const int nb = (w >> 2) * 64;

    v8f acc[4];
#pragma unroll
    for (int t = 0; t < 4; ++t)
#pragma unroll
        for (int r = 0; r < 8; ++r) acc[t][r] = 0.0f;

    // fragment addressing (ISA 7.12.2, 16-bit A 16x32 / B 32x16 layouts)
    const int kbA = (lane >> 4) * 4;   // A-frag dword base within row
    const int kbB = (lane >> 4) * 8;   // unused: B-frag is read at kbA, same k order as A
    const int am  = ms + (lane & 15);

    for (int k0 = 0; k0 < K; k0 += 32) {
        // stage A tile 64x32 (fp32 -> bf16)
#pragma unroll
        for (int it = 0; it < 2; ++it) {
            int vIdx = tid + it * 256;
            int row  = vIdx >> 3;
            int col  = (vIdx & 7) * 4;
            const float* src = A + (size_t)(m0 + row) * lda + k0 + col;
            const float4 f = *(const float4*)src;
            if (k0 + 32 < K) __builtin_prefetch(src + 32, 0, 0);   // next K-slab of stream
            unsigned short* d = &As[row * 34 + col];
            d[0] = f2bf(f.x); d[1] = f2bf(f.y); d[2] = f2bf(f.z); d[3] = f2bf(f.w);
        }
        // stage B tile 32x128 transposed (fp32 -> bf16)
#pragma unroll
        for (int it = 0; it < 4; ++it) {
            int vIdx = tid + it * 256;
            int kr   = vIdx >> 5;
            int nc   = (vIdx & 31) * 4;
            const float4 f = *(const float4*)(B + (size_t)(k0 + kr) * ldb + n0 + nc);
            Bs[(nc + 0) * 34 + kr] = f2bf(f.x);
            Bs[(nc + 1) * 34 + kr] = f2bf(f.y);
            Bs[(nc + 2) * 34 + kr] = f2bf(f.z);
            Bs[(nc + 3) * 34 + kr] = f2bf(f.w);
        }
        __syncthreads();

        FragBF a;
        const unsigned int* arow = (const unsigned int*)&As[am * 34];
#pragma unroll
        for (int d = 0; d < 4; ++d) { a.u[d] = arow[kbA + d]; a.u[4 + d] = arow[kbA + 8 + d]; }

#pragma unroll
        for (int t = 0; t < 4; ++t) {
            FragBF b;
            const unsigned int* brow =
                (const unsigned int*)&Bs[(nb + t * 16 + (lane & 15)) * 34];
#pragma unroll
            for (int d = 0; d < 4; ++d) { b.u[d] = brow[kbA + d]; b.u[4 + d] = brow[kbA + 8 + d]; }
            acc[t] = __builtin_amdgcn_wmma_f32_16x16x32_bf16(
                false, a.v, false, b.v, (short)0, acc[t], false, false);
        }
        __syncthreads();
    }

    const int ibase = m0 + ms + ((lane >> 4) << 3);
#pragma unroll
    for (int t = 0; t < 4; ++t) {
        int j = n0 + nb + t * 16 + (lane & 15);
#pragma unroll
        for (int r = 0; r < 8; ++r) {
            float vv = acc[t][r];
            if (relu) vv = fmaxf(vv, 0.0f);
            C[(size_t)(ibase + r) * ldc + j] = vv;
            if (Cbf) Cbf[(size_t)(ibase + r) * ldc + j] = f2bf(vv);
        }
    }
}

// ---------------------------------------------------------------------------
// Fused recon pass: G = Arows @ Brows^T per 128x128 tile (K = 128 in LDS).
// A/B are persistent bf16 row matrices [8192 x 128]; staged into LDS with
// async global->LDS copies (ASYNCcnt) when available.
// Consumes G in registers: weighted-CE vs adj_orig (pos_weight=1), and if
// do_stats: per-row sum(exp(sigmoid(G))) and per-row max(sigmoid(G)).
// ---------------------------------------------------------------------------
#define RSTRIDE 136   // ushorts per LDS row: 272 B, 16B-aligned, conflict-free (4m+c)%64

__global__ void __launch_bounds__(256)
recon_ce_bf16(const unsigned short* __restrict__ Abf,
              const unsigned short* __restrict__ Bbf,
              const float* __restrict__ adj_orig,
              double* __restrict__ ce_accum,
              int do_stats, float* __restrict__ sumexp, int* __restrict__ rowmaxbits)
{
    __shared__ unsigned short Ah[128 * RSTRIDE];
    __shared__ unsigned short Bh[128 * RSTRIDE];

    const int tid  = threadIdx.x;
    const int lane = tid & 31;
    const int w    = tid >> 5;
    const int i0   = blockIdx.x * 128;
    const int j0   = blockIdx.y * 128;

#if HAVE_ASYNC_LDS
    {
#pragma unroll
        for (int it = 0; it < 8; ++it) {
            int vIdx = tid + it * 256;          // 0..2047 : 128 rows x 16 chunks of 16B
            int row  = vIdx >> 4;
            int ch   = (vIdx & 15) * 16;
            const char* gA = (const char*)(Abf + (size_t)(i0 + row) * H2D) + ch;
            const char* gB = (const char*)(Bbf + (size_t)(j0 + row) * H2D) + ch;
            async_b128(gA, (char*)Ah + row * (RSTRIDE * 2) + ch);
            async_b128(gB, (char*)Bh + row * (RSTRIDE * 2) + ch);
        }
        wait_async0();
    }
#else
#pragma unroll
    for (int it = 0; it < 8; ++it) {
        int vIdx = tid + it * 256;
        int row  = vIdx >> 4;
        int ch   = (vIdx & 15) * 16;
        uint4 va = *(const uint4*)((const char*)(Abf + (size_t)(i0 + row) * H2D) + ch);
        uint4 vb = *(const uint4*)((const char*)(Bbf + (size_t)(j0 + row) * H2D) + ch);
        *(uint4*)((char*)Ah + row * (RSTRIDE * 2) + ch) = va;
        *(uint4*)((char*)Bh + row * (RSTRIDE * 2) + ch) = vb;
    }
#endif
    __syncthreads();

    v8f acc[8];
#pragma unroll
    for (int t = 0; t < 8; ++t)
#pragma unroll
        for (int r = 0; r < 8; ++r) acc[t][r] = 0.0f;

    const int kbA = (lane >> 4) * 4;
    const int kbB = (lane >> 4) * 8;
    const unsigned int* arow = (const unsigned int*)&Ah[(w * 16 + (lane & 15)) * RSTRIDE];

#pragma unroll
    for (int ks = 0; ks < 4; ++ks) {
        const int kd = ks * 16;               // dword offset of this 32-wide K slab
        FragBF a;
#pragma unroll
        for (int d = 0; d < 4; ++d) {
            a.u[d]     = arow[kd + kbA + d];
            a.u[4 + d] = arow[kd + kbA + 8 + d];
        }
#pragma unroll
        for (int t = 0; t < 8; ++t) {
            FragBF b;
            const unsigned int* brow =
                (const unsigned int*)&Bh[(t * 16 + (lane & 15)) * RSTRIDE];
#pragma unroll
            for (int d = 0; d < 4; ++d) { b.u[d] = brow[kd + kbA + d]; b.u[4 + d] = brow[kd + kbA + 8 + d]; }
            acc[t] = __builtin_amdgcn_wmma_f32_16x16x32_bf16(
                false, a.v, false, b.v, (short)0, acc[t], false, false);
        }
    }

    // ---- epilogue: consume G tile ----
    float ce_local = 0.0f;
    const int ibase = i0 + w * 16 + ((lane >> 4) << 3);
#pragma unroll
    for (int r = 0; r < 8; ++r) {
        const int i = ibase + r;
        float esum = 0.0f, smax = 0.0f;
#pragma unroll
        for (int t = 0; t < 8; ++t) {
            const float g = acc[t][r];
            const int   j = j0 + t * 16 + (lane & 15);
            const float lab = adj_orig[(size_t)i * NR + j];
            ce_local += (1.0f - lab) * g + log1pf(__expf(-fabsf(g))) + fmaxf(-g, 0.0f);
            if (do_stats) {
                const float s = 1.0f / (1.0f + __expf(-g));
                esum += __expf(s);
                smax  = fmaxf(smax, s);
            }
        }
        if (do_stats) {
#pragma unroll
            for (int off = 1; off < 16; off <<= 1) {       // reduce within 16-lane halves
                esum += __shfl_xor(esum, off, 32);
                smax  = fmaxf(smax, __shfl_xor(smax, off, 32));
            }
            if ((lane & 15) == 0) {
                atomicAdd(&sumexp[i], esum);
                atomicMax(&rowmaxbits[i], __float_as_int(smax));  // smax > 0: bit-order ok
            }
        }
    }
#pragma unroll
    for (int off = 1; off < 32; off <<= 1) ce_local += __shfl_xor(ce_local, off, 32);
    if (lane == 0) atomicAdd(ce_accum, (double)ce_local);
}

// ---------------------------------------------------------------------------
// small helper kernels
// ---------------------------------------------------------------------------
__global__ void init_accums(float* sumexp, int* rowmaxbits, double* dacc, int* maxpmibits)
{
    const int i = blockIdx.x * 256 + threadIdx.x;
    if (i < NR) { sumexp[i] = 0.0f; rowmaxbits[i] = 0; }
    if (i == 0) { for (int k = 0; k < 5; ++k) dacc[k] = 0.0; *maxpmibits = 0; }
}

__global__ void __launch_bounds__(256)
normalize_gd(const float* __restrict__ gdir, float* __restrict__ gd)
{
    const int row  = blockIdx.x * 8 + (threadIdx.x >> 5);
    const int lane = threadIdx.x & 31;
    float v[4]; float ss = 0.0f;
#pragma unroll
    for (int q = 0; q < 4; ++q) {
        v[q] = gdir[(size_t)row * H2D + lane + q * 32];
        ss += v[q] * v[q];
    }
#pragma unroll
    for (int off = 1; off < 32; off <<= 1) ss += __shfl_xor(ss, off, 32);
    const float nrm = fmaxf(sqrtf(ss), 1e-12f);
#pragma unroll
    for (int q = 0; q < 4; ++q) gd[(size_t)row * H2D + lane + q * 32] = v[q] / nrm;
}

__global__ void radius_pmi(const float* sumexp, const int* rowmaxbits,
                           float* pmi, int* maxpmibits)
{
    const int i = blockIdx.x * 256 + threadIdx.x;
    const float smax = __int_as_float(rowmaxbits[i]);
    const float p = fmaxf(0.0f, smax - __logf(sumexp[i]) + __logf(8192.0f));
    pmi[i] = p;
    atomicMax(maxpmibits, __float_as_int(p));
}

__global__ void radius_std(const float* pmi, const int* maxpmibits,
                           float* stdv, double* dacc)
{
    const int i = blockIdx.x * 256 + threadIdx.x;
    const float mp = __int_as_float(*maxpmibits);
    const float ratio = (mp > 0.0f) ? pmi[i] / mp : 0.0f;
    stdv[i] = 1.0f - ratio;
    float rs = ratio;                       // sum(1 - std) = sum(ratio)
#pragma unroll
    for (int off = 1; off < 32; off <<= 1) rs += __shfl_xor(rs, off, 32);
    if ((threadIdx.x & 31) == 0) atomicAdd(&dacc[4], (double)rs);
}

__global__ void aug_write(const float* __restrict__ h, const float* __restrict__ gd,
                          const float* __restrict__ stdv, float* __restrict__ aug,
                          unsigned short* __restrict__ aug_bf,
                          float* __restrict__ out_h, float* __restrict__ out_aug)
{
    const int idx = blockIdx.x * 256 + threadIdx.x;     // < 8192*128
    const int i   = idx >> 7;
    const float hv = h[idx];
    const float a  = hv + gd[idx] * stdv[i];
    aug[idx] = a; aug_bf[idx] = f2bf(a);
    out_h[idx] = hv; out_aug[idx] = a;
}

__global__ void __launch_bounds__(256)
instance_hinge(const float* __restrict__ aug, const float* __restrict__ h,
               const int* __restrict__ neg_idx, double* __restrict__ dacc)
{
    const int row  = blockIdx.x * 8 + (threadIdx.x >> 5);
    const int lane = threadIdx.x & 31;
    float a[4], hh[4];
    float pos = 0.0f;
#pragma unroll
    for (int q = 0; q < 4; ++q) {
        a[q]  = aug[(size_t)row * H2D + lane + q * 32];
        hh[q] = h[(size_t)row * H2D + lane + q * 32];
        pos += a[q] * hh[q];
    }
#pragma unroll
    for (int off = 1; off < 32; off <<= 1) pos += __shfl_xor(pos, off, 32);

    float ce = 0.0f, hinge = 0.0f;
    {   // positive logit, label = 1
        const float l = pos / 0.07f;
        ce += fmaxf(l, 0.0f) - l + log1pf(__expf(-fabsf(l)));
    }
    for (int k = 0; k < 10; ++k) {
        const int j = neg_idx[row * 10 + k];
        float nd = 0.0f;
#pragma unroll
        for (int q = 0; q < 4; ++q) nd += a[q] * h[(size_t)j * H2D + lane + q * 32];
#pragma unroll
        for (int off = 1; off < 32; off <<= 1) nd += __shfl_xor(nd, off, 32);
        const float l = nd / 0.07f;                     // label = 0
        ce    += fmaxf(l, 0.0f) + log1pf(__expf(-fabsf(l)));
        hinge += fmaxf(nd - pos, 0.0f);
    }
    if (lane == 0) {
        atomicAdd(&dacc[2], (double)ce);
        atomicAdd(&dacc[3], (double)hinge);
    }
}

__global__ void finalize(const double* dacc, float* out)
{
    const double NN = 8192.0 * 8192.0;
    const double gae     = 0.1 * (dacc[0] / NN);
    const double aug_gae = 0.1 * (dacc[1] / NN) * 1e-5;
    const double ins     = (dacc[2] / 8192.0) * 1e-5;
    const double hin     = (dacc[3] / 8192.0) * 0.0;
    const double nrm     = (dacc[4] / 8192.0) * (-0.1);
    const double tot = gae + aug_gae + ins + hin + nrm;
    out[0] = (float)tot; out[1] = (float)gae; out[2] = (float)aug_gae;
    out[3] = (float)ins; out[4] = (float)hin; out[5] = (float)nrm;
}

// ---------------------------------------------------------------------------
extern "C" void kernel_launch(void* const* d_in, const int* in_sizes, int n_in,
                              void* d_out, int out_size, void* d_ws, size_t ws_size,
                              hipStream_t stream)
{
    (void)in_sizes; (void)n_in; (void)out_size; (void)ws_size;

    const float* x        = (const float*)d_in[0];
    const float* adj      = (const float*)d_in[1];
    const float* adj_orig = (const float*)d_in[2];
    const float* gdir     = (const float*)d_in[3];
    const float* W1       = (const float*)d_in[4];
    const float* W2       = (const float*)d_in[5];
    const int*   neg_idx  = (const int*)d_in[6];
    float* out = (float*)d_out;

    float* wsf = (float*)d_ws;
    // region0 (8 MB): t1 -> t2 -> aug_h  |  region1 (8 MB): h1 -> h
    float* t1   = wsf;                    // 8192 x 256
    float* h1   = wsf + 2097152;          // 8192 x 256
    float* t2   = wsf;                    // 8192 x 128 (t1 dead)
    float* h    = wsf + 2097152;          // 8192 x 128 (h1 dead)
    float* aug  = wsf;                    // 8192 x 128 (t2 dead)
    float* gd   = wsf + 4194304;          // 8192 x 128
    unsigned short* h_bf16   = (unsigned short*)(wsf + 5242880);  // 8192x128 bf16 (2 MB)
    unsigned short* aug_bf16 = (unsigned short*)(wsf + 5767168);  // 8192x128 bf16 (2 MB)
    float* sumexp     = wsf + 6291456;    // 8192 f32
    int*   rowmaxbits = (int*)(wsf + 6299648);
    float* pmi  = wsf + 6307840;
    float* stdv = wsf + 6316032;
    double* dacc = (double*)(wsf + 6324224);   // 5 doubles (8B aligned)
    int* maxpmibits = (int*)(dacc + 5);

    float* out_h   = out + 6;
    float* out_aug = out + 6 + 1048576;

    init_accums <<<32,   256, 0, stream>>>(sumexp, rowmaxbits, dacc, maxpmibits);
    normalize_gd<<<1024, 256, 0, stream>>>(gdir, gd);

    // encoder: t1 = x@W1 ; h1 = relu(adj@t1) ; t2 = h1@W2 ; h = relu(adj@t2)
    gemm_nn_bf16<<<dim3(128, 2), 256, 0, stream>>>(x,   512,  W1, 256, t1, 256, 512,  0, nullptr);
    gemm_nn_bf16<<<dim3(128, 2), 256, 0, stream>>>(adj, 8192, t1, 256, h1, 256, 8192, 1, nullptr);
    gemm_nn_bf16<<<dim3(128, 1), 256, 0, stream>>>(h1,  256,  W2, 128, t2, 128, 256,  0, nullptr);
    gemm_nn_bf16<<<dim3(128, 1), 256, 0, stream>>>(adj, 8192, t2, 128, h,  128, 8192, 1, h_bf16);

    // pass 1: G = h@h^T fused -> gae CE + radius stats
    recon_ce_bf16<<<dim3(64, 64), 256, 0, stream>>>(h_bf16, h_bf16, adj_orig, &dacc[0], 1,
                                                    sumexp, rowmaxbits);
    radius_pmi<<<32, 256, 0, stream>>>(sumexp, rowmaxbits, pmi, maxpmibits);
    radius_std<<<32, 256, 0, stream>>>(pmi, maxpmibits, stdv, dacc);
    aug_write <<<4096, 256, 0, stream>>>(h, gd, stdv, aug, aug_bf16, out_h, out_aug);

    // pass 2: aug_h@h^T fused -> aug CE
    recon_ce_bf16<<<dim3(64, 64), 256, 0, stream>>>(aug_bf16, h_bf16, adj_orig, &dacc[1], 0,
                                                    sumexp, rowmaxbits);

    instance_hinge<<<1024, 256, 0, stream>>>(aug, h, neg_idx, dacc);
    finalize<<<1, 1, 0, stream>>>(dacc, out);
}
